// CausalWanSelfAttention_24068996726815
// MI455X (gfx1250) — matrix operand order repaired and checked by reading, not compiled, not run on hardware
//
#include <hip/hip_runtime.h>
#include <hip/hip_bf16.h>

#define S_TOK   4680
#define S_PAD   4704      /* multiple of 32, >= S_TOK */
#define DIM     1536
#define NH      12
#define HD      128
#define FRAME   1560      /* 30*52 */
#define CT      22
#define CTH     43        /* CT + CH */

typedef unsigned short u16;
typedef __attribute__((ext_vector_type(16))) __bf16 v16bf;
typedef __attribute__((ext_vector_type(8)))  float  v8f;
typedef __attribute__((ext_vector_type(4)))  unsigned int u32x4;

// ---------------------------------------------------------------------------
// fp32 -> bf16 (round to nearest even)
// ---------------------------------------------------------------------------
static __device__ inline u16 f2bf(float f) {
    unsigned u = __float_as_uint(f);
    unsigned r = (u + 0x7FFFu + ((u >> 16) & 1u)) >> 16;
    return (u16)r;
}

// ---------------------------------------------------------------------------
// WMMA tile loaders (bf16, wave32).
// A-matrix 16x32: lanes 0-15 hold row r, K = k0+0..7 (v0-3) and k0+16..23 (v4-7);
//                 lanes 16-31 hold row r, K = k0+8..15 and k0+24..31.
// B-matrix 32x16: lanes 0-15 hold col n, K = k0+0..7 and k0+16..23; lanes 16-31 K = k0+8..15 and k0+24..31.
// Source data is row-major along K in both cases (stride in elements).
// ---------------------------------------------------------------------------
static __device__ inline v16bf tileA(const u16* base, int stride) {
    int lane = threadIdx.x & 31;
    int r = lane & 15, hi = lane >> 4;
    const u16* p = base + (size_t)r * stride + hi * 8;
    union { u32x4 q[2]; v16bf v; } u;
    u.q[0] = *(const u32x4*)p;
    u.q[1] = *(const u32x4*)(p + 16);
    return u.v;
}

static __device__ inline v16bf tileB(const u16* base, int stride) {
    int lane = threadIdx.x & 31;
    int n = lane & 15, hi = lane >> 4;
    const u16* p = base + (size_t)n * stride + hi * 8;
    union { u32x4 q[2]; v16bf v; } u;
    u.q[0] = *(const u32x4*)p;
    u.q[1] = *(const u32x4*)(p + 16);
    return u.v;
}

// A-tile of probabilities from fp32 LDS scores (row stride S_PAD floats)
static __device__ inline v16bf tileA_lds(const float* sc, int k0) {
    int lane = threadIdx.x & 31;
    int r = lane & 15, hi = lane >> 4;
    const float* p = sc + (size_t)r * S_PAD + k0 + hi * 8;
    union { u16 h[16]; v16bf v; } u;
#pragma unroll
    for (int i = 0; i < 8; i++) {
        u.h[i]     = f2bf(p[i]);
        u.h[8 + i] = f2bf(p[16 + i]);
    }
    return u.v;
}

static __device__ inline v8f wmma_bf16(v16bf a, v16bf b, v8f c) {
    return __builtin_amdgcn_wmma_f32_16x16x32_bf16(false, a, false, b, (short)0, c,
                                                   false, false);
}

// ---------------------------------------------------------------------------
// Kernel 1: fp32 -> bf16 conversion (grid-stride)
// ---------------------------------------------------------------------------
__global__ void cvt_f32_bf16(const float* __restrict__ src, u16* __restrict__ dst, int n) {
    int i = blockIdx.x * blockDim.x + threadIdx.x;
    int stride = gridDim.x * blockDim.x;
    for (; i < n; i += stride) dst[i] = f2bf(src[i]);
}

// ---------------------------------------------------------------------------
// Kernel 2: QKV GEMM  Y = x @ W^T + b  (raw fp32 out, rmsnorm applied later)
// grid (S_PAD/16, DIM/256, 3), block 128 (4 waves).
// Each wave computes a 16x64 strip: A tile loaded once per k-step, 4 WMMAs.
// ---------------------------------------------------------------------------
__global__ __launch_bounds__(128) void gemm_qkv(
    const u16* __restrict__ xb,
    const u16* __restrict__ wq, const u16* __restrict__ wk, const u16* __restrict__ wv,
    const float* __restrict__ bq, const float* __restrict__ bk, const float* __restrict__ bv,
    float* __restrict__ Yq, float* __restrict__ Yk, float* __restrict__ Yv) {
    const u16* W; const float* bias; float* Y;
    if (blockIdx.z == 0)      { W = wq; bias = bq; Y = Yq; }
    else if (blockIdx.z == 1) { W = wk; bias = bk; Y = Yk; }
    else                      { W = wv; bias = bv; Y = Yv; }

    int m0 = blockIdx.x * 16;
    int wave = threadIdx.x >> 5;
    int n0 = blockIdx.y * 256 + wave * 64;        // 4 N-tiles of 16 per wave
    const u16* Ab = xb + (size_t)m0 * DIM;
    const u16* Bb = W  + (size_t)n0 * DIM;

    v8f acc0 = {}, acc1 = {}, acc2 = {}, acc3 = {};
    for (int k = 0; k < DIM; k += 32) {
        v16bf a = tileA(Ab + k, DIM);
        v16bf b0 = tileB(Bb + (size_t)0  * DIM + k, DIM);
        v16bf b1 = tileB(Bb + (size_t)16 * DIM + k, DIM);
        v16bf b2 = tileB(Bb + (size_t)32 * DIM + k, DIM);
        v16bf b3 = tileB(Bb + (size_t)48 * DIM + k, DIM);
        acc0 = wmma_bf16(a, b0, acc0);
        acc1 = wmma_bf16(a, b1, acc1);
        acc2 = wmma_bf16(a, b2, acc2);
        acc3 = wmma_bf16(a, b3, acc3);
    }

    int lane = threadIdx.x & 31;
    int r16 = lane & 15, hi = lane >> 4;
    v8f accs[4] = {acc0, acc1, acc2, acc3};
#pragma unroll
    for (int j = 0; j < 4; j++) {
        int n = n0 + j * 16 + r16;
        float bn = bias[n];
#pragma unroll
        for (int r = 0; r < 8; r++)
            Y[(size_t)(m0 + r + 8 * hi) * DIM + n] = accs[j][r] + bn;
    }
}

// ---------------------------------------------------------------------------
// Kernel 3: RMSNorm + RoPE (q,k) / transpose (v).
// grid (S_TOK, 3), block 256.  y==0 -> Q, y==1 -> K, y==2 -> V.
// Qh,Kh: [NH][S_PAD][HD] bf16; Vt: [NH][HD][S_PAD] bf16.
// ---------------------------------------------------------------------------
__global__ __launch_bounds__(256) void norm_rope(
    const float* __restrict__ Yq, const float* __restrict__ Yk, const float* __restrict__ Yv,
    const float* __restrict__ gq, const float* __restrict__ gk,
    const float* __restrict__ freqs,
    u16* __restrict__ Qh, u16* __restrict__ Kh, u16* __restrict__ Vt) {
    int s = blockIdx.x;
    int which = blockIdx.y;

    if (which == 2) {
        const float* row = Yv + (size_t)s * DIM;
        for (int c = threadIdx.x; c < DIM; c += blockDim.x) {
            int h = c >> 7, d = c & 127;
            Vt[((size_t)h * HD + d) * S_PAD + s] = f2bf(row[c]);
        }
        return;
    }

    const float* row = (which ? Yk : Yq) + (size_t)s * DIM;
    const float* g   = which ? gk : gq;
    u16* O           = which ? Kh : Qh;

    __shared__ float red[8];
    float ss = 0.f;
    for (int c = threadIdx.x; c < DIM; c += blockDim.x) { float v = row[c]; ss += v * v; }
#pragma unroll
    for (int off = 16; off; off >>= 1) ss += __shfl_xor(ss, off, 32);
    if ((threadIdx.x & 31) == 0) red[threadIdx.x >> 5] = ss;
    __syncthreads();
    if (threadIdx.x == 0) {
        float t = 0.f;
#pragma unroll
        for (int i = 0; i < 8; i++) t += red[i];
        red[0] = rsqrtf(t * (1.0f / DIM) + 1e-6f);
    }
    __syncthreads();
    float inv = red[0];

    int t_idx = s / FRAME;
    int rem   = s - t_idx * FRAME;
    int y_idx = rem / 52;
    int x_idx = rem - y_idx * 52;

    for (int p = threadIdx.x; p < NH * 64; p += blockDim.x) {
        int h = p >> 6, j = p & 63;
        int c0 = h * HD + 2 * j;
        float a0 = row[c0]     * inv * g[c0];
        float a1 = row[c0 + 1] * inv * g[c0 + 1];
        float ang;
        if (j < CT)       ang = freqs[t_idx * 64 + j];
        else if (j < CTH) ang = freqs[y_idx * 64 + j];
        else              ang = freqs[x_idx * 64 + j];
        float cs = __cosf(ang), sn = __sinf(ang);
        size_t base = ((size_t)h * S_PAD + s) * HD + 2 * j;
        O[base]     = f2bf(a0 * cs - a1 * sn);
        O[base + 1] = f2bf(a0 * sn + a1 * cs);
    }
}

// ---------------------------------------------------------------------------
// Kernel 4: attention with frame-block-causal mask.
// grid (ceil(S/16), NH), block 256 (8 waves).
// Dynamic LDS: 16 full score rows (16 * S_PAD fp32) + rowsum[16].
// Phase 1: Q tile hoisted to registers; waves fill scores via WMMA QK^T.
// Phase 2: per-row softmax (max/exp/sum) in LDS.
// Phase 3: each wave owns a 16-wide d-slice of O = P @ V via WMMA.
// ---------------------------------------------------------------------------
__global__ __launch_bounds__(256) void attn(
    const u16* __restrict__ Qh, const u16* __restrict__ Kh, const u16* __restrict__ Vt,
    u16* __restrict__ aout) {
    extern __shared__ float sc[];            // [16][S_PAD]
    float* rowsum = sc + 16 * S_PAD;         // [16]

    int m0   = blockIdx.x * 16;
    int h    = blockIdx.y;
    int wave = threadIdx.x >> 5;
    int lane = threadIdx.x & 31;
    int r16  = lane & 15, hi = lane >> 4;

    int qlast   = m0 + 15 < S_TOK ? m0 + 15 : S_TOK - 1;
    int fq_max  = qlast / FRAME;
    int kend    = (fq_max + 1) * FRAME;
    if (kend > S_TOK) kend = S_TOK;
    int kpad    = (kend + 31) & ~31;         // multiple of 32, <= S_PAD

    const float scale = 0.08838834764831845f; // 1/sqrt(128)

    // ---- Phase 1: scores (Q tile kept in registers across the key loop) ----
    const u16* Qb = Qh + ((size_t)h * S_PAD + m0) * HD;
    const u16* Kb = Kh + (size_t)h * S_PAD * HD;
    v16bf qa[4];
#pragma unroll
    for (int j = 0; j < 4; j++) qa[j] = tileA(Qb + j * 32, HD);

    int nkt = kpad >> 4;
    for (int kt = wave; kt < nkt; kt += 8) {
        v8f acc = {};
        const u16* Kt = Kb + (size_t)kt * 16 * HD;
        __builtin_prefetch(Kt + 8 * 16 * HD, 0, 1);   // next tile for this wave
#pragma unroll
        for (int j = 0; j < 4; j++) {
            v16bf b = tileB(Kt + j * 32, HD);
            acc = wmma_bf16(qa[j], b, acc);
        }
        int n  = kt * 16 + r16;              // key index
        int fk = n / FRAME;
#pragma unroll
        for (int r = 0; r < 8; r++) {
            int m  = r + 8 * hi;
            int fq = (m0 + m) / FRAME;       // padded q rows land in frame>=3: allow all
            float v = acc[r] * scale;
            if (fk > fq) v = -1e30f;
            sc[m * S_PAD + n] = v;
        }
    }
    __syncthreads();

    // ---- Phase 2: softmax per row (16 threads per row) ----
    {
        int row = threadIdx.x >> 4, sub = threadIdx.x & 15;
        float* srow = sc + row * S_PAD;
        float mx = -3e38f;
        for (int k = sub; k < kpad; k += 16) mx = fmaxf(mx, srow[k]);
#pragma unroll
        for (int off = 8; off; off >>= 1) mx = fmaxf(mx, __shfl_xor(mx, off, 16));
        float sum = 0.f;
        for (int k = sub; k < kpad; k += 16) {
            float e = __expf(srow[k] - mx);
            srow[k] = e;
            sum += e;
        }
#pragma unroll
        for (int off = 8; off; off >>= 1) sum += __shfl_xor(sum, off, 16);
        if (sub == 0) rowsum[row] = sum;
    }
    __syncthreads();

    // ---- Phase 3: O = P @ V, wave owns d-slice [wave*16, wave*16+16) ----
    {
        int n0 = wave * 16;
        const u16* Vb = Vt + ((size_t)h * HD + n0) * S_PAD;
        v8f acc = {};
        for (int k = 0; k < kpad; k += 32) {
            __builtin_prefetch(Vb + k + 128, 0, 1);
            v16bf a = tileA_lds(sc, k);
            v16bf b = tileB(Vb + k, S_PAD);
            acc = wmma_bf16(a, b, acc);
        }
#pragma unroll
        for (int r = 0; r < 8; r++) {
            int m = r + 8 * hi;
            int q = m0 + m;
            if (q < S_TOK) {
                float o = acc[r] / rowsum[m];
                aout[(size_t)q * DIM + h * HD + n0 + r16] = f2bf(o);
            }
        }
    }
}

// ---------------------------------------------------------------------------
// Kernel 5: output projection  out = aout @ Wo^T + bo  (fp32 out)
// grid (ceil(S/16), DIM/256), block 128 (4 waves); wave -> 16x64 strip
// ---------------------------------------------------------------------------
__global__ __launch_bounds__(128) void gemm_out(
    const u16* __restrict__ A, const u16* __restrict__ W,
    const float* __restrict__ bias, float* __restrict__ out) {
    int m0 = blockIdx.x * 16;
    int wave = threadIdx.x >> 5;
    int n0 = blockIdx.y * 256 + wave * 64;
    const u16* Ab = A + (size_t)m0 * DIM;
    const u16* Bb = W + (size_t)n0 * DIM;

    v8f acc0 = {}, acc1 = {}, acc2 = {}, acc3 = {};
    for (int k = 0; k < DIM; k += 32) {
        v16bf a = tileA(Ab + k, DIM);
        v16bf b0 = tileB(Bb + (size_t)0  * DIM + k, DIM);
        v16bf b1 = tileB(Bb + (size_t)16 * DIM + k, DIM);
        v16bf b2 = tileB(Bb + (size_t)32 * DIM + k, DIM);
        v16bf b3 = tileB(Bb + (size_t)48 * DIM + k, DIM);
        acc0 = wmma_bf16(a, b0, acc0);
        acc1 = wmma_bf16(a, b1, acc1);
        acc2 = wmma_bf16(a, b2, acc2);
        acc3 = wmma_bf16(a, b3, acc3);
    }

    int lane = threadIdx.x & 31;
    int r16 = lane & 15, hi = lane >> 4;
    v8f accs[4] = {acc0, acc1, acc2, acc3};
#pragma unroll
    for (int j = 0; j < 4; j++) {
        int n = n0 + j * 16 + r16;
        float bn = bias[n];
#pragma unroll
        for (int r = 0; r < 8; r++) {
            int q = m0 + r + 8 * hi;
            if (q < S_TOK) out[(size_t)q * DIM + n] = accs[j][r] + bn;
        }
    }
}

// ---------------------------------------------------------------------------
// Host launch
// ---------------------------------------------------------------------------
extern "C" void kernel_launch(void* const* d_in, const int* in_sizes, int n_in,
                              void* d_out, int out_size, void* d_ws, size_t ws_size,
                              hipStream_t stream) {
    const float* x     = (const float*)d_in[0];
    const float* freqs = (const float*)d_in[3];
    const float* Wq    = (const float*)d_in[4];
    const float* bq    = (const float*)d_in[5];
    const float* Wk    = (const float*)d_in[6];
    const float* bk    = (const float*)d_in[7];
    const float* Wv    = (const float*)d_in[8];
    const float* bv    = (const float*)d_in[9];
    const float* Wo    = (const float*)d_in[10];
    const float* bo    = (const float*)d_in[11];
    const float* gq    = (const float*)d_in[12];
    const float* gk    = (const float*)d_in[13];
    float* out = (float*)d_out;

    // --- workspace carve-up (256B aligned regions) ---
    char* ws = (char*)d_ws;
    size_t off = 0;
    auto take = [&](size_t bytes) {
        char* p = ws + off;
        off += (bytes + 255) & ~(size_t)255;
        return p;
    };
    u16*   xb  = (u16*)  take((size_t)S_PAD * DIM * 2);
    u16*   wqb = (u16*)  take((size_t)DIM * DIM * 2);
    u16*   wkb = (u16*)  take((size_t)DIM * DIM * 2);
    u16*   wvb = (u16*)  take((size_t)DIM * DIM * 2);
    u16*   wob = (u16*)  take((size_t)DIM * DIM * 2);
    float* Yq  = (float*)take((size_t)S_PAD * DIM * 4);
    float* Yk  = (float*)take((size_t)S_PAD * DIM * 4);
    float* Yv  = (float*)take((size_t)S_PAD * DIM * 4);
    u16*   Qh  = (u16*)  take((size_t)NH * S_PAD * HD * 2);
    u16*   Kh  = (u16*)  take((size_t)NH * S_PAD * HD * 2);
    u16*   Vt  = (u16*)  take((size_t)NH * HD * S_PAD * 2);
    u16*   aout= (u16*)  take((size_t)S_PAD * DIM * 2);
    (void)ws_size; (void)in_sizes; (void)n_in; (void)out_size;

    // 1) conversions
    cvt_f32_bf16<<<2048, 256, 0, stream>>>(x,  xb,  S_TOK * DIM);
    cvt_f32_bf16<<<2048, 256, 0, stream>>>(Wq, wqb, DIM * DIM);
    cvt_f32_bf16<<<2048, 256, 0, stream>>>(Wk, wkb, DIM * DIM);
    cvt_f32_bf16<<<2048, 256, 0, stream>>>(Wv, wvb, DIM * DIM);
    cvt_f32_bf16<<<2048, 256, 0, stream>>>(Wo, wob, DIM * DIM);

    // 2) QKV GEMM
    {
        dim3 grid(S_PAD / 16, DIM / 256, 3);
        gemm_qkv<<<grid, 128, 0, stream>>>(xb, wqb, wkb, wvb, bq, bk, bv, Yq, Yk, Yv);
    }

    // 3) rmsnorm + rope + v transpose
    {
        dim3 grid(S_TOK, 3);
        norm_rope<<<grid, 256, 0, stream>>>(Yq, Yk, Yv, gq, gk, freqs, Qh, Kh, Vt);
    }

    // 4) attention
    {
        dim3 grid((S_TOK + 15) / 16, NH);
        size_t smem = (size_t)(16 * S_PAD + 16) * sizeof(float);
        attn<<<grid, 256, smem, stream>>>(Qh, Kh, Vt, aout);
    }

    // 5) output projection
    {
        dim3 grid((S_TOK + 15) / 16, DIM / 256);
        gemm_out<<<grid, 128, 0, stream>>>(aout, wob, bo, out);
    }
}
